// MultiHeadAttention_34342558499177
// MI455X (gfx1250) — matrix operand order repaired and checked by reading, not compiled, not run on hardware
//
#include <hip/hip_runtime.h>
#include <hip/hip_bf16.h>

// ---------------------------------------------------------------------------
// MHA forward for MI455X (gfx1250, wave32, WMMA).
// Pipeline: f32->f16 convert, 3x WMMA projection GEMMs (head-split epilogues),
// flash-attention (online softmax, no score materialization), WMMA out-proj.
// Wave tiles enlarged (GEMM 32x64, flash 32 queries) to amortize B/K/V frag
// loads over 2x the WMMA issue -> compute-bound inner loops.
// ---------------------------------------------------------------------------

typedef __attribute__((ext_vector_type(16))) _Float16 v16h;
typedef __attribute__((ext_vector_type(8)))  _Float16 v8h;
typedef __attribute__((ext_vector_type(4)))  _Float16 v4h;
typedef __attribute__((ext_vector_type(8)))  float    v8f;

constexpr int BATCH = 4;
constexpr int SEQ   = 2048;
constexpr int DM    = 1024;
constexpr int NH    = 16;
constexpr int DH    = 64;
constexpr int MTOT  = BATCH * SEQ;   // 8192

__device__ __forceinline__ v8f vzero8() {
    v8f z;
#pragma unroll
    for (int i = 0; i < 8; ++i) z[i] = 0.0f;
    return z;
}

__device__ __forceinline__ v16h pack16(v8h lo, v8h hi) {
    v16h r;
#pragma unroll
    for (int i = 0; i < 8; ++i) { r[i] = lo[i]; r[i + 8] = hi[i]; }
    return r;
}

__device__ __forceinline__ v8f wmma_f16(v16h a, v16h b, v8f c) {
    return __builtin_amdgcn_wmma_f32_16x16x32_f16(false, a, false, b,
                                                  (short)0, c, false, false);
}

// ---------------------------------------------------------------------------
// f32 -> f16 conversion (vectorized x4). n is a multiple of 1024.
// ---------------------------------------------------------------------------
__global__ __launch_bounds__(256) void cvt_f32_f16(const float* __restrict__ src,
                                                   _Float16* __restrict__ dst,
                                                   int n) {
    int i = (blockIdx.x * 256 + threadIdx.x) * 4;
    if (i < n) {
        float4 f = *(const float4*)(src + i);
        v4h h;
        h[0] = (_Float16)f.x; h[1] = (_Float16)f.y;
        h[2] = (_Float16)f.z; h[3] = (_Float16)f.w;
        *(v4h*)(dst + i) = h;
    }
}

// ---------------------------------------------------------------------------
// Y = X(f16, MxK) @ W(f16, NxK)^T   (nn.Linear semantics, y = x @ W.T)
// Block: 256 threads = 8 waves; wave tile 32(M) x 64(N); K step 32.
// Per k-step: 2 A-frag loads + 4 B-frag loads feed 8 WMMA.
// mode 0: f16, [b,h,s,dh] layout, *scale   (Qh / Kh)
// mode 2: f16, [b,h,dh,s] layout            (Vt, transposed for PV B-frags)
// mode 3: f32, row-major [M,N]              (final output)
// ---------------------------------------------------------------------------
__global__ __launch_bounds__(256) void gemm_xwt(const _Float16* __restrict__ X,
                                                const _Float16* __restrict__ W,
                                                void* __restrict__ Y,
                                                int K, int N, float scale, int mode) {
    const int lane = threadIdx.x & 31;
    const int wave = threadIdx.x >> 5;
    const int m0 = (blockIdx.x * 8 + wave) * 32;
    const int n0 = blockIdx.y * 64;

    // A-fragment addressing (16x32 f16): lane m = L&15, 16 contiguous K at kq
    const int kq = 16 * (lane >> 4);
    // B-fragment addressing (32x16 f16): lane n = L&15, 16 contiguous K at 16*(L>>4)
    const int bn = lane & 15;
    const int kb = 16 * (lane >> 4);

    v8f acc[2][4];
#pragma unroll
    for (int mi = 0; mi < 2; ++mi)
#pragma unroll
        for (int t = 0; t < 4; ++t) acc[mi][t] = vzero8();

    const _Float16* aRow0 = X + (size_t)(m0 + (lane & 15)) * K;
    const _Float16* aRow1 = aRow0 + (size_t)16 * K;
    for (int k0 = 0; k0 < K; k0 += 32) {
        const _Float16* ap0 = aRow0 + k0 + kq;
        const _Float16* ap1 = aRow1 + k0 + kq;
        v16h a0 = pack16(*(const v8h*)ap0, *(const v8h*)(ap0 + 8));
        v16h a1 = pack16(*(const v8h*)ap1, *(const v8h*)(ap1 + 8));
        __builtin_prefetch(ap0 + 32, 0, 0);  // next K tile -> global_prefetch_b8
        __builtin_prefetch(ap1 + 32, 0, 0);
#pragma unroll
        for (int t = 0; t < 4; ++t) {
            const _Float16* bp = W + (size_t)(n0 + t * 16 + bn) * K + k0 + kb;
            v16h b = *(const v16h*)bp;
            acc[0][t] = wmma_f16(a0, b, acc[0][t]);
            acc[1][t] = wmma_f16(a1, b, acc[1][t]);
        }
    }

    // C/D layout: lane holds column n = L&15, rows r + 8*(L>>4)
    const int rbase = 8 * (lane >> 4);
#pragma unroll
    for (int mi = 0; mi < 2; ++mi) {
#pragma unroll
        for (int t = 0; t < 4; ++t) {
            const int n = n0 + t * 16 + (lane & 15);
#pragma unroll
            for (int r = 0; r < 8; ++r) {
                const int m = m0 + mi * 16 + rbase + r;
                const float v = acc[mi][t][r] * scale;
                if (mode == 3) {
                    ((float*)Y)[(size_t)m * N + n] = v;
                } else {
                    const _Float16 hv = (_Float16)v;
                    const int b = m >> 11, s = m & (SEQ - 1);
                    const int h = n >> 6, d = n & (DH - 1);
                    if (mode == 2)  // Vt [b,h,dh,s]
                        ((_Float16*)Y)[((size_t)(b * NH + h) * DH + d) * SEQ + s] = hv;
                    else            // Qh / Kh [b,h,s,dh]
                        ((_Float16*)Y)[((size_t)(b * NH + h) * SEQ + s) * DH + d] = hv;
                }
            }
        }
    }
}

// ---------------------------------------------------------------------------
// Flash attention: grid (B*H, SEQ/256), 256 threads = 8 waves.
// Wave owns 32 query rows (two 16-row tiles); streams 32 keys/iter.
// K/V fragments shared across both query tiles -> 16 WMMA per 8 frag loads.
// Q is pre-scaled by 1/sqrt(dh) in its projection epilogue.
// ---------------------------------------------------------------------------
__global__ __launch_bounds__(256) void flash_attn(const _Float16* __restrict__ Qh,
                                                  const _Float16* __restrict__ Kh,
                                                  const _Float16* __restrict__ Vt,
                                                  _Float16* __restrict__ Ctx) {
    __shared__ _Float16 lds[8][2][16 * 32];  // per-wave, per-q-tile P staging

    const int lane = threadIdx.x & 31;
    const int wave = threadIdx.x >> 5;
    const int bh = blockIdx.x;              // b*NH + h
    const int b = bh >> 4, h = bh & 15;
    const int q0 = (blockIdx.y * 8 + wave) * 32;

    const _Float16* Qb = Qh + (size_t)bh * SEQ * DH;
    const _Float16* Kb = Kh + (size_t)bh * SEQ * DH;
    const _Float16* Vb = Vt + (size_t)bh * DH * SEQ;

    const int kq = 16 * (lane >> 4);        // A-frag contiguous-K offset
    const int nl = lane & 15;               // B/C-frag column lane
    const int kb = 16 * (lane >> 4);        // B-frag contiguous-K offset
    const int hi = lane >> 4;               // row-group selector for C layout

    // Q fragments, resident for the whole K loop: [q-tile][dh chunk]
    v16h qa[2][2];
#pragma unroll
    for (int qt = 0; qt < 2; ++qt)
#pragma unroll
        for (int c = 0; c < 2; ++c) {
            const _Float16* qp =
                Qb + (size_t)(q0 + qt * 16 + (lane & 15)) * DH + c * 32 + kq;
            qa[qt][c] = pack16(*(const v8h*)qp, *(const v8h*)(qp + 8));
        }

    v8f acco[2][4];                         // per q-tile: 16q x 64dh, 4 col chunks
#pragma unroll
    for (int qt = 0; qt < 2; ++qt)
#pragma unroll
        for (int c = 0; c < 4; ++c) acco[qt][c] = vzero8();
    float mrow[2][8], lrow[2][8];
#pragma unroll
    for (int qt = 0; qt < 2; ++qt)
#pragma unroll
        for (int r = 0; r < 8; ++r) { mrow[qt][r] = -1.0e30f; lrow[qt][r] = 0.0f; }

    for (int kt = 0; kt < SEQ; kt += 32) {
        // ---- scores: 2 q-tiles x 2 key tiles, K-dim = dh = 64 (2 chunks) ----
        v8f s[2][2];
#pragma unroll
        for (int qt = 0; qt < 2; ++qt) { s[qt][0] = vzero8(); s[qt][1] = vzero8(); }
#pragma unroll
        for (int c = 0; c < 2; ++c) {
            v16h bk0 = *(const v16h*)(Kb + (size_t)(kt + nl) * DH + c * 32 + kb);
            v16h bk1 = *(const v16h*)(Kb + (size_t)(kt + 16 + nl) * DH + c * 32 + kb);
#pragma unroll
            for (int qt = 0; qt < 2; ++qt) {
                s[qt][0] = wmma_f16(qa[qt][c], bk0, s[qt][0]);
                s[qt][1] = wmma_f16(qa[qt][c], bk1, s[qt][1]);
            }
        }

        // ---- online softmax over the 32 new columns, per q-tile, per row ----
        float alpha[2][8];
#pragma unroll
        for (int qt = 0; qt < 2; ++qt) {
            _Float16* myP = &lds[wave][qt][0];
#pragma unroll
            for (int r = 0; r < 8; ++r) {
                float mx = fmaxf(s[qt][0][r], s[qt][1][r]);
                mx = fmaxf(mx, __shfl_xor(mx, 1, 16));
                mx = fmaxf(mx, __shfl_xor(mx, 2, 16));
                mx = fmaxf(mx, __shfl_xor(mx, 4, 16));
                mx = fmaxf(mx, __shfl_xor(mx, 8, 16));
                const float mnew = fmaxf(mrow[qt][r], mx);
                alpha[qt][r] = __expf(mrow[qt][r] - mnew);
                const float p0 = __expf(s[qt][0][r] - mnew);
                const float p1 = __expf(s[qt][1][r] - mnew);
                float ps = p0 + p1;
                ps += __shfl_xor(ps, 1, 16);
                ps += __shfl_xor(ps, 2, 16);
                ps += __shfl_xor(ps, 4, 16);
                ps += __shfl_xor(ps, 8, 16);
                lrow[qt][r] = lrow[qt][r] * alpha[qt][r] + ps;
                mrow[qt][r] = mnew;
                // stage P in LDS: row = r + 8*(L>>4), cols {nl, nl+16}
                const int mloc = r + 8 * hi;
                myP[mloc * 32 + nl]      = (_Float16)p0;
                myP[mloc * 32 + 16 + nl] = (_Float16)p1;
            }
        }

        // same-wave LDS RAW: C-layout -> A-layout re-swizzle round trip
        asm volatile("s_wait_dscnt 0" ::: "memory");

        v16h pf[2];
#pragma unroll
        for (int qt = 0; qt < 2; ++qt) {
            const _Float16* pp = &lds[wave][qt][0] + (lane & 15) * 32 + kq;
            pf[qt] = pack16(*(const v8h*)pp, *(const v8h*)(pp + 8));
        }

        // ---- rescale and accumulate P(16x32) x V(32x64), V shared ----
#pragma unroll
        for (int c = 0; c < 4; ++c) {
            v16h vb = *(const v16h*)(Vb + (size_t)(c * 16 + nl) * SEQ + kt + kb);
#pragma unroll
            for (int qt = 0; qt < 2; ++qt) {
#pragma unroll
                for (int r = 0; r < 8; ++r) acco[qt][c][r] *= alpha[qt][r];
                acco[qt][c] = wmma_f16(pf[qt], vb, acco[qt][c]);
            }
        }
    }

    // ---- normalize and store merged-head context [b, s, h*dh] (f16) ----
#pragma unroll
    for (int qt = 0; qt < 2; ++qt)
#pragma unroll
        for (int c = 0; c < 4; ++c)
#pragma unroll
            for (int r = 0; r < 8; ++r) {
                const int m = q0 + qt * 16 + r + 8 * hi;
                const int d = h * DH + c * 16 + nl;
                Ctx[((size_t)b * SEQ + m) * DM + d] =
                    (_Float16)(acco[qt][c][r] / lrow[qt][r]);
            }
}

// ---------------------------------------------------------------------------
extern "C" void kernel_launch(void* const* d_in, const int* in_sizes, int n_in,
                              void* d_out, int out_size, void* d_ws, size_t ws_size,
                              hipStream_t stream) {
    const float* q  = (const float*)d_in[0];
    const float* k  = (const float*)d_in[1];
    const float* v  = (const float*)d_in[2];
    const float* Wq = (const float*)d_in[3];
    const float* Wk = (const float*)d_in[4];
    const float* Wv = (const float*)d_in[5];
    const float* Wo = (const float*)d_in[6];

    const size_t nAct = (size_t)MTOT * DM;  // 8,388,608
    const size_t nW   = (size_t)DM * DM;    // 1,048,576

    _Float16* p = (_Float16*)d_ws;
    _Float16* q16  = p; p += nAct;
    _Float16* k16  = p; p += nAct;
    _Float16* v16  = p; p += nAct;
    _Float16* Wq16 = p; p += nW;
    _Float16* Wk16 = p; p += nW;
    _Float16* Wv16 = p; p += nW;
    _Float16* Wo16 = p; p += nW;
    _Float16* Qh   = p; p += nAct;
    _Float16* Kh   = p; p += nAct;
    _Float16* Vt   = p; p += nAct;
    _Float16* Ctx  = p; p += nAct;

    const int actBlocks = (int)(nAct / 1024);  // 4 elems/thread, 256 threads
    const int wBlocks   = (int)(nW / 1024);
    cvt_f32_f16<<<actBlocks, 256, 0, stream>>>(q, q16, (int)nAct);
    cvt_f32_f16<<<actBlocks, 256, 0, stream>>>(k, k16, (int)nAct);
    cvt_f32_f16<<<actBlocks, 256, 0, stream>>>(v, v16, (int)nAct);
    cvt_f32_f16<<<wBlocks, 256, 0, stream>>>(Wq, Wq16, (int)nW);
    cvt_f32_f16<<<wBlocks, 256, 0, stream>>>(Wk, Wk16, (int)nW);
    cvt_f32_f16<<<wBlocks, 256, 0, stream>>>(Wv, Wv16, (int)nW);
    cvt_f32_f16<<<wBlocks, 256, 0, stream>>>(Wo, Wo16, (int)nW);

    dim3 gg(MTOT / 256, DM / 64);  // (32, 16); wave tile 32x64
    // Q projection: fold softmax 1/sqrt(dh) into Q
    gemm_xwt<<<gg, 256, 0, stream>>>(q16, Wq16, Qh, DM, DM, 0.125f, 0);
    gemm_xwt<<<gg, 256, 0, stream>>>(k16, Wk16, Kh, DM, DM, 1.0f, 0);
    gemm_xwt<<<gg, 256, 0, stream>>>(v16, Wv16, Vt, DM, DM, 1.0f, 2);

    flash_attn<<<dim3(BATCH * NH, SEQ / 256), 256, 0, stream>>>(Qh, Kh, Vt, Ctx);

    gemm_xwt<<<gg, 256, 0, stream>>>(Ctx, Wo16, d_out, DM, DM, 1.0f, 3);
}
